// PointNetPlusPlus_81733227643071
// MI455X (gfx1250) — matrix operand order repaired and checked by reading, not compiled, not run on hardware
//
#include <hip/hip_runtime.h>
#include <hip/hip_bf16.h>

// ---------------------------------------------------------------------------
// PointNet++ semantic-seg forward for MI455X (gfx1250, wave32, WMMA).
// GEMM layers use v_wmma_f32_16x16x32_f16 (f16 operands, f32 accum): each
// layer is followed by training-mode BatchNorm which renormalizes
// activations, so f16 range is safe and f32 accumulation preserves BN stats.
// Fragments are loaded straight from global memory with b128 loads in the
// exact ISA 7.12.2 register layout (A: two 16B runs/lane; B column-major:
// two 16B runs/lane) -- no LDS staging, no barriers in the inner loop.
// K dims are zero-padded to multiples of 32.
// ---------------------------------------------------------------------------

typedef __attribute__((ext_vector_type(16))) _Float16 v16h;
typedef __attribute__((ext_vector_type(8)))  float    v8f;

#ifndef FLT_BIG
#define FLT_BIG 3.4e38f
#endif

// ============================ FPS ==========================================
__global__ __launch_bounds__(256)
void fps_kernel(const float* __restrict__ xyz, float* __restrict__ new_xyz,
                int N, int K)
{
    __shared__ float mind[8192];
    __shared__ float rmax[256];
    __shared__ int   rarg[256];
    __shared__ int   s_cur;

    const int b   = blockIdx.x;
    const int tid = threadIdx.x;
    const float* pts = xyz + (size_t)b * N * 3;

    for (int i = tid; i < N; i += 256) mind[i] = FLT_BIG;
    if (tid == 0) s_cur = 0;
    __syncthreads();

    for (int it = 0; it < K; ++it) {
        const int cur = s_cur;
        const float cx = pts[cur*3+0], cy = pts[cur*3+1], cz = pts[cur*3+2];
        if (tid == 0) {
            float* o = new_xyz + ((size_t)b * K + it) * 3;
            o[0] = cx; o[1] = cy; o[2] = cz;
        }
        float best = -1.0f; int barg = 0;
        for (int i = tid; i < N; i += 256) {
            float dx = pts[i*3+0] - cx, dy = pts[i*3+1] - cy, dz = pts[i*3+2] - cz;
            float d  = dx*dx + dy*dy + dz*dz;
            float m  = fminf(mind[i], d);
            mind[i] = m;
            if (m > best) { best = m; barg = i; }
        }
        rmax[tid] = best; rarg[tid] = barg;
        __syncthreads();
        for (int s = 128; s > 0; s >>= 1) {
            if (tid < s) {
                float om = rmax[tid+s]; int oa = rarg[tid+s];
                if (om > rmax[tid] || (om == rmax[tid] && oa < rarg[tid])) {
                    rmax[tid] = om; rarg[tid] = oa;
                }
            }
            __syncthreads();
        }
        if (tid == 0) s_cur = rarg[0];
        __syncthreads();
    }
}

// ========================= Ball query ======================================
__global__ __launch_bounds__(128)
void ball_query_kernel(const float* __restrict__ new_xyz,
                       const float* __restrict__ xyz,
                       int* __restrict__ out_idx,
                       int P, int N, int K, float r2)
{
    __shared__ int s_cnt;
    __shared__ int s_wcnt[4];
    __shared__ int s_woff[4];

    const int tid  = threadIdx.x;
    const int lane = tid & 31;
    const int wid  = tid >> 5;
    const int q    = blockIdx.x;          // b*P + p
    const int b    = q / P;
    const float* pts = xyz + (size_t)b * N * 3;
    const float qx = new_xyz[(size_t)q*3+0];
    const float qy = new_xyz[(size_t)q*3+1];
    const float qz = new_xyz[(size_t)q*3+2];
    int* out = out_idx + (size_t)q * K;

    if (tid == 0) s_cnt = 0;
    __syncthreads();

    for (int base = 0; base < N; base += 128) {
        const int i = base + tid;
        bool within = false;
        if (i < N) {
            float dx = pts[i*3+0]-qx, dy = pts[i*3+1]-qy, dz = pts[i*3+2]-qz;
            within = (dx*dx + dy*dy + dz*dz) < r2;
        }
        unsigned m32 = (unsigned)__ballot(within);
        int prefix = __popc(m32 & ((1u << lane) - 1u));
        if (lane == 0) s_wcnt[wid] = __popc(m32);
        __syncthreads();
        if (tid == 0) {
            int run = s_cnt;
            for (int w = 0; w < 4; ++w) { s_woff[w] = run; run += s_wcnt[w]; }
            s_cnt = run;
        }
        __syncthreads();
        if (within) {
            int slot = s_woff[wid] + prefix;
            if (slot < K) out[slot] = i;
        }
        __syncthreads();
        if (s_cnt >= K) break;
    }
    int cnt = s_cnt; if (cnt > K) cnt = K;
    for (int j = cnt + tid; j < K; j += 128) out[j] = N - 1;
}

// ==================== Grouping / concat builders ===========================
__global__ void group_build_kernel(const float* __restrict__ xyz,      // (B,N,3)
                                   const float* __restrict__ new_xyz,  // (B,P,3)
                                   const _Float16* __restrict__ feats, // (B,Cf,N) or null
                                   int Cf,
                                   const int* __restrict__ idx,        // (B,P,K) or null
                                   _Float16* __restrict__ A,
                                   int Kpad, int N, int P, int K, size_t total)
{
    size_t t = (size_t)blockIdx.x * blockDim.x + threadIdx.x;
    if (t >= total) return;
    int c = (int)(t % Kpad);
    size_t pix = t / Kpad;
    int k  = (int)(pix % K);
    int pp = (int)((pix / K) % P);
    int b  = (int)(pix / ((size_t)K * P));
    int j  = idx ? idx[pix] : k;
    float v = 0.0f;
    if (c < 3) {
        v = xyz[((size_t)b * N + j) * 3 + c] - new_xyz[((size_t)b * P + pp) * 3 + c];
    } else if (c < 3 + Cf) {
        v = (float)feats[((size_t)b * Cf + (c - 3)) * N + j];
    }
    A[t] = (_Float16)v;
}

__global__ void centroid_kernel(const float* __restrict__ xyz, float* __restrict__ out,
                                int N, int total)
{
    int t = blockIdx.x * blockDim.x + threadIdx.x;   // B*3 threads
    if (t >= total) return;
    int b = t / 3, d = t % 3;
    float s = 0.0f;
    for (int j = 0; j < N; ++j) s += xyz[((size_t)b * N + j) * 3 + d];
    out[b * 3 + d] = s / (float)N;
}

// ====================== 3-NN + interpolation ===============================
__global__ void knn3_kernel(const float* __restrict__ q_xyz,   // (B,Nq,3)
                            const float* __restrict__ r_xyz,   // (B,Nr,3)
                            int* __restrict__ idx3, float* __restrict__ w3,
                            int Nq, int Nr, size_t total)
{
    size_t t = (size_t)blockIdx.x * blockDim.x + threadIdx.x;
    if (t >= total) return;
    int b = (int)(t / Nq);
    const float qx = q_xyz[t*3+0], qy = q_xyz[t*3+1], qz = q_xyz[t*3+2];
    const float* ref = r_xyz + (size_t)b * Nr * 3;
    float d0 = FLT_BIG, d1 = FLT_BIG, d2 = FLT_BIG;
    int   i0 = 0, i1 = -1, i2 = -1;
    for (int j = 0; j < Nr; ++j) {
        float dx = ref[j*3+0]-qx, dy = ref[j*3+1]-qy, dz = ref[j*3+2]-qz;
        float d = dx*dx + dy*dy + dz*dz;
        if (d < d0)      { d2=d1; i2=i1; d1=d0; i1=i0; d0=d; i0=j; }
        else if (d < d1) { d2=d1; i2=i1; d1=d;  i1=j; }
        else if (d < d2) { d2=d;  i2=j; }
    }
    if (i1 < 0) { i1 = i0; d1 = d0; }      // keff<3 -> repeat nearest
    if (i2 < 0) { i2 = i0; d2 = d0; }
    d0 = fmaxf(d0, 1e-10f); d1 = fmaxf(d1, 1e-10f); d2 = fmaxf(d2, 1e-10f);
    float w0 = 1.0f/d0, w1 = 1.0f/d1, w2 = 1.0f/d2;
    float s  = w0 + w1 + w2;
    idx3[t*3+0] = i0; idx3[t*3+1] = i1; idx3[t*3+2] = i2;
    w3[t*3+0] = w0/s; w3[t*3+1] = w1/s; w3[t*3+2] = w2/s;
}

__global__ void interp_build_kernel(const _Float16* __restrict__ coarsef, // (B,Cc,Nr)
                                    int Cc, int Nr,
                                    const _Float16* __restrict__ finef,   // (B,Cf,Nq) or null
                                    int Cf,
                                    const int* __restrict__ idx3,
                                    const float* __restrict__ w3,
                                    _Float16* __restrict__ A,
                                    int Kpad, int Nq, size_t total)
{
    size_t t = (size_t)blockIdx.x * blockDim.x + threadIdx.x;
    if (t >= total) return;
    int c = (int)(t % Kpad);
    size_t pix = t / Kpad;
    int b = (int)(pix / Nq), p = (int)(pix % Nq);
    float v = 0.0f;
    if (c < Cc) {
        const int*   id = idx3 + pix * 3;
        const float* ww = w3   + pix * 3;
        const _Float16* f = coarsef + ((size_t)b * Cc + c) * Nr;
        v = ww[0]*(float)f[id[0]] + ww[1]*(float)f[id[1]] + ww[2]*(float)f[id[2]];
    } else if (c < Cc + Cf) {
        v = (float)finef[((size_t)b * Cf + (c - Cc)) * Nq + p];
    }
    A[t] = (_Float16)v;
}

// =============== Weight pad (column-major) + stat reset ====================
// Wc[n*Kpad + k] = w[n,k]  (each output channel's K-vector contiguous, so a
// WMMA B-fragment is two contiguous 16B runs per lane).
__global__ void wconv_kernel(const float* __restrict__ w, _Float16* __restrict__ Wc,
                             float* __restrict__ cs, float* __restrict__ cq,
                             int Cout, int Cin, int Kpad)
{
    int t = blockIdx.x * blockDim.x + threadIdx.x;
    if (t < Cout) { cs[t] = 0.0f; cq[t] = 0.0f; }
    if (t >= Kpad * Cout) return;
    int n = t / Kpad, k = t % Kpad;
    Wc[t] = (k < Cin) ? (_Float16)w[(size_t)n * Cin + k] : (_Float16)0.0f;
}

// ============================ WMMA GEMM ====================================
// Block = 256 threads = 8 wave32 waves. Each wave computes a 16x32 D tile
// (2 WMMAs per K-step sharing one A fragment); block covers 128 rows x 32
// cols. Fragments loaded directly from global in WMMA register layout:
//   A (row-major, stride Ka):  lane(m,g) -> row m, halves [g*8..+7] and
//                              [16+g*8..+7] of the K-tile  (2x b128)
//   B (column-major, stride Ka): lane(n,g) -> column n, halves [g*8..+7]
//                              and [16+g*8..+7] of the K-tile (2x b128)
__global__ __launch_bounds__(256)
void gemm_kernel(const _Float16* __restrict__ A, const _Float16* __restrict__ Wc,
                 const float* __restrict__ bias, _Float16* __restrict__ Y,
                 float* __restrict__ chanSum, float* __restrict__ chanSqs,
                 int M, int Ka, int Nc)
{
    const int tid  = threadIdx.x;
    const int wid  = tid >> 5;
    const int lane = tid & 31;
    const int m    = lane & 15;        // A row within tile / B column within tile
    const int g    = lane >> 4;        // half-wave group

    const int rowA  = blockIdx.x * 128 + wid * 16 + m;
    const int nbase = blockIdx.y * 32;

    const _Float16* arow  = A  + (size_t)rowA * Ka;
    const _Float16* bcol0 = Wc + (size_t)(nbase +      m) * Ka;
    const _Float16* bcol1 = Wc + (size_t)(nbase + 16 + m) * Ka;

    v8f acc0 = {0.f,0.f,0.f,0.f,0.f,0.f,0.f,0.f};
    v8f acc1 = {0.f,0.f,0.f,0.f,0.f,0.f,0.f,0.f};

    const int kTiles = Ka >> 5;
    for (int kt = 0; kt < kTiles; ++kt) {
        const int kb = kt * 32;
        v16h af, bf0, bf1;
        ((int4*)&af )[0] = *(const int4*)(arow  + kb +      g * 8);
        ((int4*)&af )[1] = *(const int4*)(arow  + kb + 16 + g * 8);
        ((int4*)&bf0)[0] = *(const int4*)(bcol0 + kb +      g * 8);
        ((int4*)&bf0)[1] = *(const int4*)(bcol0 + kb + 16 + g * 8);
        ((int4*)&bf1)[0] = *(const int4*)(bcol1 + kb +      g * 8);
        ((int4*)&bf1)[1] = *(const int4*)(bcol1 + kb + 16 + g * 8);
        if (kt + 1 < kTiles)   // gfx1250 global_prefetch_b8 for next A tile
            __builtin_prefetch(arow + kb + 32, 0, 1);
        acc0 = __builtin_amdgcn_wmma_f32_16x16x32_f16(
                   false, af, false, bf0, (short)0, acc0, false, false);
        acc1 = __builtin_amdgcn_wmma_f32_16x16x32_f16(
                   false, af, false, bf1, (short)0, acc1, false, false);
    }

    // Epilogue: bias, f16 store, BN statistics.
    // D layout: lane(n,g) holds N=n, M=g*8+r for VGPR r.
    const int n      = lane & 15;
    const int col0   = nbase + n;
    const int col1   = col0 + 16;
    const int rowOut = blockIdx.x * 128 + wid * 16 + g * 8;
    const float bv0 = bias[col0];
    const float bv1 = bias[col1];
    float s0 = 0.f, q0 = 0.f, s1 = 0.f, q1 = 0.f;
#pragma unroll
    for (int r = 0; r < 8; ++r) {
        float v0 = acc0[r] + bv0;
        float v1 = acc1[r] + bv1;
        s0 += v0; q0 += v0 * v0;
        s1 += v1; q1 += v1 * v1;
        size_t base = (size_t)(rowOut + r) * Nc;
        Y[base + col0] = (_Float16)v0;
        Y[base + col1] = (_Float16)v1;
    }
    atomicAdd(&chanSum[col0], s0);
    atomicAdd(&chanSqs[col0], q0);
    atomicAdd(&chanSum[col1], s1);
    atomicAdd(&chanSqs[col1], q1);
}

// ========================= BatchNorm stages ================================
__global__ void bn_finalize_kernel(const float* __restrict__ cs, const float* __restrict__ cq,
                                   float* __restrict__ mean, float* __restrict__ rs,
                                   int Nc, float invM)
{
    int c = blockIdx.x * blockDim.x + threadIdx.x;
    if (c >= Nc) return;
    float m = cs[c] * invM;
    float v = cq[c] * invM - m * m;
    mean[c] = m;
    rs[c]   = rsqrtf(v + 1e-5f);
}

__global__ void bn_apply_kernel(_Float16* __restrict__ Y,
                                const float* __restrict__ mean, const float* __restrict__ rs,
                                const float* __restrict__ gamma, const float* __restrict__ beta,
                                size_t total, int Nc)
{
    size_t i = (size_t)blockIdx.x * blockDim.x + threadIdx.x;
    if (i >= total) return;
    int c = (int)(i % Nc);
    float v = (float)Y[i];
    v = (v - mean[c]) * rs[c] * gamma[c] + beta[c];
    Y[i] = (_Float16)(v > 0.0f ? v : 0.0f);
}

__global__ void bn_apply_out_kernel(const _Float16* __restrict__ Y,
                                    const float* __restrict__ mean, const float* __restrict__ rs,
                                    const float* __restrict__ gamma, const float* __restrict__ beta,
                                    float* __restrict__ out, int Nfine, int Nc, size_t total)
{
    size_t i = (size_t)blockIdx.x * blockDim.x + threadIdx.x;
    if (i >= total) return;
    int c = (int)(i % Nc);
    size_t row = i / Nc;
    int b = (int)(row / Nfine), n = (int)(row % Nfine);
    float v = (float)Y[i];
    v = (v - mean[c]) * rs[c] * gamma[c] + beta[c];
    v = v > 0.0f ? v : 0.0f;
    out[((size_t)b * Nc + c) * Nfine + n] = v;
}

// ===================== Max-pool over nsample, transpose ====================
__global__ void maxpool_kernel(const _Float16* __restrict__ Y, _Float16* __restrict__ feat,
                               int P, int K, int C, size_t total)   // total = B*C*P
{
    size_t t = (size_t)blockIdx.x * blockDim.x + threadIdx.x;
    if (t >= total) return;
    int b   = (int)(t / ((size_t)C * P));
    int rem = (int)(t % ((size_t)C * P));
    int c = rem / P, p = rem % P;
    float m = -FLT_BIG;
    const _Float16* base = Y + ((size_t)(b * P + p) * K) * C + c;
    for (int k = 0; k < K; ++k) {
        float v = (float)base[(size_t)k * C];
        m = fmaxf(m, v);
    }
    feat[t] = (_Float16)m;
}

__global__ void rows_to_bcp_kernel(const _Float16* __restrict__ Y, _Float16* __restrict__ feat,
                                   int P, int C, size_t total)       // total = B*C*P
{
    size_t t = (size_t)blockIdx.x * blockDim.x + threadIdx.x;
    if (t >= total) return;
    int b   = (int)(t / ((size_t)C * P));
    int rem = (int)(t % ((size_t)C * P));
    int c = rem / P, p = rem % P;
    feat[t] = Y[((size_t)b * P + p) * C + c];
}

// ============================== Host side ==================================
struct LayerP { const float *w, *b, *gamma, *beta; int Cin, Cout, Kpad; };

static inline dim3 gridFor(size_t total, int bs = 256) {
    return dim3((unsigned)((total + bs - 1) / bs));
}

extern "C" void kernel_launch(void* const* d_in, const int* in_sizes, int n_in,
                              void* d_out, int out_size, void* d_ws, size_t ws_size,
                              hipStream_t stream)
{
    const int B = 16, N0 = 8192, P1 = 512, K1 = 32, P2 = 128, K2 = 64;

    // Canonical layer order: sa1(3), sa2(3), sa3(3), fp3(2), fp2(2), fp1(3)
    static const int cins[16]  = {3,64,64, 131,128,128, 259,256,512, 1280,256, 384,256, 128,128,128};
    static const int couts[16] = {64,64,128, 128,128,256, 256,512,1024, 256,256, 256,128, 128,128,128};

    int pcIdx = -1;
    for (int i = 0; i < n_in; ++i)
        if (in_sizes[i] == B * N0 * 3) { pcIdx = i; break; }
    if (pcIdx < 0) pcIdx = 0;
    const float* pc = (const float*)d_in[pcIdx];

    LayerP L[16];
    if (pcIdx == 0) {
        // dict-insertion flattening: pointcloud, sa1..fp1; each layer w,b,gamma,beta
        for (int g = 0; g < 16; ++g) {
            L[g].w     = (const float*)d_in[1 + g*4 + 0];
            L[g].b     = (const float*)d_in[1 + g*4 + 1];
            L[g].gamma = (const float*)d_in[1 + g*4 + 2];
            L[g].beta  = (const float*)d_in[1 + g*4 + 3];
        }
    } else {
        // sorted tree flattening: params first (fp1,fp2,fp3,sa1,sa2,sa3; b,beta,gamma,w)
        static const int ord[16] = {13,14,15, 11,12, 9,10, 0,1,2, 3,4,5, 6,7,8};
        for (int g = 0; g < 16; ++g) {
            int ci = ord[g];
            L[ci].b     = (const float*)d_in[g*4 + 0];
            L[ci].beta  = (const float*)d_in[g*4 + 1];
            L[ci].gamma = (const float*)d_in[g*4 + 2];
            L[ci].w     = (const float*)d_in[g*4 + 3];
        }
    }
    for (int i = 0; i < 16; ++i) {
        L[i].Cin = cins[i]; L[i].Cout = couts[i];
        L[i].Kpad = ((cins[i] + 31) / 32) * 32;
    }

    // ---- workspace bump allocator ----
    char* ws = (char*)d_ws;
    size_t off = 0;
    auto alloc = [&](size_t bytes) -> void* {
        off = (off + 255) & ~(size_t)255;
        void* p = ws + off;
        off += bytes;
        return p;
    };
    float*    l1_xyz  = (float*)alloc((size_t)B*P1*3*4);
    float*    l2_xyz  = (float*)alloc((size_t)B*P2*3*4);
    float*    l3_xyz  = (float*)alloc((size_t)B*3*4);
    int*      idx1    = (int*)  alloc((size_t)B*P1*K1*4);
    int*      idx2    = (int*)  alloc((size_t)B*P2*K2*4);
    int*      knn_idx = (int*)  alloc((size_t)B*N0*3*4);
    float*    knn_w   = (float*)alloc((size_t)B*N0*3*4);
    _Float16* l1_f    = (_Float16*)alloc((size_t)B*128*P1*2);
    _Float16* l2_f    = (_Float16*)alloc((size_t)B*256*P2*2);
    _Float16* l3_f    = (_Float16*)alloc((size_t)B*1024*2);
    _Float16* fp3_f   = (_Float16*)alloc((size_t)B*256*P2*2);
    _Float16* fp2_f   = (_Float16*)alloc((size_t)B*128*P1*2);
    _Float16* Wc      = (_Float16*)alloc((size_t)1280*1024*2);
    float*    chanSum = (float*)alloc(1024*4);
    float*    chanSqs = (float*)alloc(1024*4);
    float*    meanb   = (float*)alloc(1024*4);
    float*    rsb     = (float*)alloc(1024*4);
    _Float16* ping    = (_Float16*)alloc((size_t)64*1024*1024);
    _Float16* pong    = (_Float16*)alloc((size_t)64*1024*1024);

    auto run_layer = [&](const LayerP& lp, const _Float16* A, _Float16* Y,
                         int M, bool final_out) {
        int tot_w = lp.Kpad * lp.Cout;
        wconv_kernel<<<gridFor((size_t)tot_w), 256, 0, stream>>>(
            lp.w, Wc, chanSum, chanSqs, lp.Cout, lp.Cin, lp.Kpad);
        dim3 g(M / 128, lp.Cout / 32);
        gemm_kernel<<<g, 256, 0, stream>>>(A, Wc, lp.b, Y, chanSum, chanSqs,
                                           M, lp.Kpad, lp.Cout);
        bn_finalize_kernel<<<gridFor((size_t)lp.Cout), 256, 0, stream>>>(
            chanSum, chanSqs, meanb, rsb, lp.Cout, 1.0f / (float)M);
        size_t total = (size_t)M * lp.Cout;
        if (!final_out)
            bn_apply_kernel<<<gridFor(total), 256, 0, stream>>>(
                Y, meanb, rsb, lp.gamma, lp.beta, total, lp.Cout);
        else
            bn_apply_out_kernel<<<gridFor(total), 256, 0, stream>>>(
                Y, meanb, rsb, lp.gamma, lp.beta, (float*)d_out, N0, lp.Cout, total);
    };

    // ================= SA1: FPS->512, ball(r=0.2,K=32), MLP [3,64,64,128] ===
    fps_kernel<<<B, 256, 0, stream>>>(pc, l1_xyz, N0, P1);
    ball_query_kernel<<<B*P1, 128, 0, stream>>>(l1_xyz, pc, idx1, P1, N0, K1, 0.04f);
    {
        int M = B * P1 * K1;                       // 262144
        size_t tot = (size_t)M * 32;
        group_build_kernel<<<gridFor(tot), 256, 0, stream>>>(
            pc, l1_xyz, nullptr, 0, idx1, ping, 32, N0, P1, K1, tot);
        _Float16 *Ab = ping, *Yb = pong;
        for (int li = 0; li < 3; ++li) {
            run_layer(L[li], Ab, Yb, M, false);
            _Float16* t = Ab; Ab = Yb; Yb = t;
        }
        maxpool_kernel<<<gridFor((size_t)B*128*P1), 256, 0, stream>>>(
            Ab, l1_f, P1, K1, 128, (size_t)B*128*P1);
    }

    // ================= SA2: FPS->128, ball(r=0.4,K=64), MLP [131,...] ======
    fps_kernel<<<B, 256, 0, stream>>>(l1_xyz, l2_xyz, P1, P2);
    ball_query_kernel<<<B*P2, 128, 0, stream>>>(l2_xyz, l1_xyz, idx2, P2, P1, K2, 0.16f);
    {
        int M = B * P2 * K2;                       // 131072
        size_t tot = (size_t)M * 160;
        group_build_kernel<<<gridFor(tot), 256, 0, stream>>>(
            l1_xyz, l2_xyz, l1_f, 128, idx2, ping, 160, P1, P2, K2, tot);
        _Float16 *Ab = ping, *Yb = pong;
        for (int li = 3; li < 6; ++li) {
            run_layer(L[li], Ab, Yb, M, false);
            _Float16* t = Ab; Ab = Yb; Yb = t;
        }
        maxpool_kernel<<<gridFor((size_t)B*256*P2), 256, 0, stream>>>(
            Ab, l2_f, P2, K2, 256, (size_t)B*256*P2);
    }

    // ================= SA3: group_all over 128 pts, MLP [259,...] ==========
    centroid_kernel<<<gridFor((size_t)B*3, 64), 64, 0, stream>>>(l2_xyz, l3_xyz, P2, B*3);
    {
        int M = B * P2;                            // 2048 (P=1, K=128)
        size_t tot = (size_t)M * 288;
        group_build_kernel<<<gridFor(tot), 256, 0, stream>>>(
            l2_xyz, l3_xyz, l2_f, 256, nullptr, ping, 288, P2, 1, P2, tot);
        _Float16 *Ab = ping, *Yb = pong;
        for (int li = 6; li < 9; ++li) {
            run_layer(L[li], Ab, Yb, M, false);
            _Float16* t = Ab; Ab = Yb; Yb = t;
        }
        maxpool_kernel<<<gridFor((size_t)B*1024), 256, 0, stream>>>(
            Ab, l3_f, 1, P2, 1024, (size_t)B*1024);
    }

    // ================= FP3: interp l3->l2 + concat l2_f, MLP [1280,256,256] =
    knn3_kernel<<<gridFor((size_t)B*P2), 256, 0, stream>>>(
        l2_xyz, l3_xyz, knn_idx, knn_w, P2, 1, (size_t)B*P2);
    {
        int M = B * P2;
        size_t tot = (size_t)M * 1280;
        interp_build_kernel<<<gridFor(tot), 256, 0, stream>>>(
            l3_f, 1024, 1, l2_f, 256, knn_idx, knn_w, ping, 1280, P2, tot);
        _Float16 *Ab = ping, *Yb = pong;
        for (int li = 9; li < 11; ++li) {
            run_layer(L[li], Ab, Yb, M, false);
            _Float16* t = Ab; Ab = Yb; Yb = t;
        }
        rows_to_bcp_kernel<<<gridFor((size_t)B*256*P2), 256, 0, stream>>>(
            Ab, fp3_f, P2, 256, (size_t)B*256*P2);
    }

    // ================= FP2: interp l2->l1 + concat l1_f, MLP [384,256,128] ==
    knn3_kernel<<<gridFor((size_t)B*P1), 256, 0, stream>>>(
        l1_xyz, l2_xyz, knn_idx, knn_w, P1, P2, (size_t)B*P1);
    {
        int M = B * P1;                            // 8192
        size_t tot = (size_t)M * 384;
        interp_build_kernel<<<gridFor(tot), 256, 0, stream>>>(
            fp3_f, 256, P2, l1_f, 128, knn_idx, knn_w, ping, 384, P1, tot);
        _Float16 *Ab = ping, *Yb = pong;
        for (int li = 11; li < 13; ++li) {
            run_layer(L[li], Ab, Yb, M, false);
            _Float16* t = Ab; Ab = Yb; Yb = t;
        }
        rows_to_bcp_kernel<<<gridFor((size_t)B*128*P1), 256, 0, stream>>>(
            Ab, fp2_f, P1, 128, (size_t)B*128*P1);
    }

    // ================= FP1: interp l1->xyz, MLP [128,128,128,128] -> d_out ==
    knn3_kernel<<<gridFor((size_t)B*N0), 256, 0, stream>>>(
        pc, l1_xyz, knn_idx, knn_w, N0, P1, (size_t)B*N0);
    {
        int M = B * N0;                            // 131072
        size_t tot = (size_t)M * 128;
        interp_build_kernel<<<gridFor(tot), 256, 0, stream>>>(
            fp2_f, 128, P1, nullptr, 0, knn_idx, knn_w, ping, 128, N0, tot);
        _Float16 *Ab = ping, *Yb = pong;
        for (int li = 13; li < 16; ++li) {
            run_layer(L[li], Ab, Yb, M, li == 15);
            _Float16* t = Ab; Ab = Yb; Yb = t;
        }
    }
    (void)in_sizes; (void)out_size; (void)ws_size;
}
